// DistributedAttention_12833362280569
// MI455X (gfx1250) — matrix operand order repaired and checked by reading, not compiled, not run on hardware
//
#include <hip/hip_runtime.h>

// Problem constants
#define B_   8
#define T_   1024
#define C_   512
#define H_   8
#define D_   64
#define SCALE_ 0.125f          // 64^-0.5
#define BRSEG 4194304u         // B_*H_*T_*D_ = B_*T_*C_ halves per branch segment

typedef __attribute__((ext_vector_type(16))) _Float16 v16h;
typedef __attribute__((ext_vector_type(8)))  _Float16 v8h;
typedef __attribute__((ext_vector_type(8)))  float    v8f;

__device__ __forceinline__ v8f wmma16(v16h a, v16h b, v8f c) {
  // D = A(16x32 f16) * B(32x16 f16) + C(16x16 f32)
  return __builtin_amdgcn_wmma_f32_16x16x32_f16(false, a, false, b, (short)0, c,
                                                false, false);
}

// ---------------------------------------------------------------------------
// Fragment loaders implementing the CDNA5 ISA VGPR layouts (CDNA5 ISA):
//  A (16x32 f16): lane holds row m = lane&15; K chunks [koff,koff+8) and
//                 [koff+16,koff+24) with koff = 8*(lane>>4).
//  B (32x16 f16): loaded from B^T (N x K row-major): lane holds col n=lane&15,
//                 K chunks [koff,koff+8) and [koff+16,koff+24), koff = 8*(lane>>4).
// ---------------------------------------------------------------------------
__device__ __forceinline__ v16h frag_a_f16(const _Float16* base, int ld, int k0, int lane) {
  int r    = lane & 15;
  int koff = (lane >> 4) << 3;
  const _Float16* p = base + (size_t)r * ld + k0 + koff;
  v8h lo = *(const v8h*)(p);
  v8h hi = *(const v8h*)(p + 16);
  return __builtin_shufflevector(lo, hi, 0,1,2,3,4,5,6,7,8,9,10,11,12,13,14,15);
}

__device__ __forceinline__ v16h frag_b_f16(const _Float16* baseT, int ld, int k0, int lane) {
  int r    = lane & 15;
  int koff = (lane >> 4) << 3;
  const _Float16* p = baseT + (size_t)r * ld + k0 + koff;
  v8h lo = *(const v8h*)(p);
  v8h hi = *(const v8h*)(p + 16);
  return __builtin_shufflevector(lo, hi, 0,1,2,3,4,5,6,7,8,9,10,11,12,13,14,15);
}

__device__ __forceinline__ v16h frag_a_f32(const float* base, int ld, int k0, int lane) {
  int r    = lane & 15;
  int koff = (lane >> 4) << 3;
  const float* p = base + (size_t)r * ld + k0 + koff;
  float4 a0 = ((const float4*)p)[0];
  float4 a1 = ((const float4*)p)[1];
  float4 b0 = ((const float4*)(p + 16))[0];
  float4 b1 = ((const float4*)(p + 16))[1];
  v16h f;
  f[0]=(_Float16)a0.x; f[1]=(_Float16)a0.y; f[2]=(_Float16)a0.z; f[3]=(_Float16)a0.w;
  f[4]=(_Float16)a1.x; f[5]=(_Float16)a1.y; f[6]=(_Float16)a1.z; f[7]=(_Float16)a1.w;
  f[8]=(_Float16)b0.x; f[9]=(_Float16)b0.y; f[10]=(_Float16)b0.z; f[11]=(_Float16)b0.w;
  f[12]=(_Float16)b1.x; f[13]=(_Float16)b1.y; f[14]=(_Float16)b1.z; f[15]=(_Float16)b1.w;
  return f;
}

__device__ __forceinline__ v16h frag_b_f32(const float* baseT, int ld, int k0, int lane) {
  int r    = lane & 15;
  int koff = (lane >> 4) << 3;
  const float* p = baseT + (size_t)r * ld + k0 + koff;
  float4 c0 = ((const float4*)p)[0];
  float4 c1 = ((const float4*)p)[1];
  float4 c2 = ((const float4*)(p + 16))[0];
  float4 c3 = ((const float4*)(p + 16))[1];
  v16h f;
  f[0]=(_Float16)c0.x; f[1]=(_Float16)c0.y; f[2]=(_Float16)c0.z; f[3]=(_Float16)c0.w;
  f[4]=(_Float16)c1.x; f[5]=(_Float16)c1.y; f[6]=(_Float16)c1.z; f[7]=(_Float16)c1.w;
  f[8]=(_Float16)c2.x; f[9]=(_Float16)c2.y; f[10]=(_Float16)c2.z; f[11]=(_Float16)c2.w;
  f[12]=(_Float16)c3.x; f[13]=(_Float16)c3.y; f[14]=(_Float16)c3.z; f[15]=(_Float16)c3.w;
  return f;
}

struct WArgs { const float* w[9]; };                 // [br*3 + {q,k,v}]
struct OArgs { const float* wo[3]; const float* bo[3]; };

// ---------------------------------------------------------------------------
// Kernel 1: QKV projections. C[m,n] = sum_k X[m, br*512+k] * W[n,k]
// One wave per 32x32 output tile (2x2 fragment blocking: 4 WMMA share 4 frags).
// Q/K stored [b][h][t][d] f16; V stored transposed [b][h][d][t] f16.
// ---------------------------------------------------------------------------
__global__ void qkv_kernel(const float* __restrict__ x, WArgs wa,
                           _Float16* __restrict__ Qh, _Float16* __restrict__ Kh,
                           _Float16* __restrict__ Vt) {
  int wave = threadIdx.x >> 5, lane = threadIdx.x & 31;
  int tile = blockIdx.x * (blockDim.x >> 5) + wave;   // 0..4095
  int mt = tile >> 4;                                 // 256 M-tiles (32 rows)
  int nt = tile & 15;                                 // 16  N-tiles (32 cols)
  int y  = blockIdx.y;
  int br = y / 3, sel = y % 3;

  const float* A0 = x + (size_t)mt * 32 * (3 * C_) + br * C_;
  const float* A1 = A0 + (size_t)16 * (3 * C_);
  const float* W0 = wa.w[y] + (size_t)nt * 32 * C_;
  const float* W1 = W0 + (size_t)16 * C_;

  v8f cc[2][2] = {{{}, {}}, {{}, {}}};
  #pragma unroll 2
  for (int k0 = 0; k0 < C_; k0 += 32) {
    v16h a0 = frag_a_f32(A0, 3 * C_, k0, lane);
    v16h a1 = frag_a_f32(A1, 3 * C_, k0, lane);
    v16h b0 = frag_b_f32(W0, C_, k0, lane);
    v16h b1 = frag_b_f32(W1, C_, k0, lane);
    cc[0][0] = wmma16(a0, b0, cc[0][0]);
    cc[0][1] = wmma16(a0, b1, cc[0][1]);
    cc[1][0] = wmma16(a1, b0, cc[1][0]);
    cc[1][1] = wmma16(a1, b1, cc[1][1]);
  }

  _Float16* out = (sel == 0 ? Qh : (sel == 1 ? Kh : Vt)) + (size_t)br * BRSEG;
  int hl = lane >> 4;
  #pragma unroll
  for (int mi = 0; mi < 2; ++mi) {
    #pragma unroll
    for (int ni = 0; ni < 2; ++ni) {
      int n = nt * 32 + ni * 16 + (lane & 15);
      int h = n >> 6, d = n & 63;
      #pragma unroll
      for (int j = 0; j < 8; ++j) {
        int m = mt * 32 + mi * 16 + j + (hl << 3);
        int b_ = m >> 10, t = m & 1023;
        size_t idx = (sel == 2)
            ? (((size_t)(b_ * H_ + h) * D_ + d) * T_ + t)     // V^T: [b][h][d][t]
            : (((size_t)(b_ * H_ + h) * T_ + t) * D_ + d);    // Q/K: [b][h][t][d]
        out[idx] = (_Float16)cc[mi][ni][j];
      }
    }
  }
}

// ---------------------------------------------------------------------------
// Kernel 2: flash attention per (br, b, h, 16-row q-tile), one wave each.
// Block = 4 waves; per-wave 16x40 f16 LDS tile to relayout P (C-frag -> A-frag).
// ---------------------------------------------------------------------------
__global__ void attn_kernel(const _Float16* __restrict__ Qh,
                            const _Float16* __restrict__ Kh,
                            const _Float16* __restrict__ Vt,
                            _Float16* __restrict__ Oh) {
  __shared__ _Float16 smem[4][16][40];
  int wave = threadIdx.x >> 5, lane = threadIdx.x & 31;
  int qt = blockIdx.x * 4 + wave;          // 0..63
  int bh = blockIdx.y;                     // b*H_ + h, 0..63
  int br = blockIdx.z;                     // 0..2
  size_t head = (size_t)br * 64 + bh;      // (br*B_+b)*H_+h

  const _Float16* Qb = Qh + head * (T_ * D_) + (size_t)qt * 16 * D_;
  const _Float16* Kb = Kh + head * (T_ * D_);
  const _Float16* Vb = Vt + head * (D_ * T_);

  v16h aq0 = frag_a_f16(Qb, D_, 0, lane);
  v16h aq1 = frag_a_f16(Qb, D_, 32, lane);

  float mrun[8], lrun[8];
  v8f o[4] = {{}, {}, {}, {}};
  #pragma unroll
  for (int j = 0; j < 8; ++j) { mrun[j] = -1e30f; lrun[j] = 0.0f; }

  int hl = lane >> 4, col = lane & 15;

  for (int kk = 0; kk < T_; kk += 32) {
    // S = Q(16x64) * K^T : two 16-key column groups, K-dim 64 in 2 steps
    v8f s0 = {}, s1 = {};
    s0 = wmma16(aq0, frag_b_f16(Kb + (size_t)kk * D_, D_, 0, lane), s0);
    s0 = wmma16(aq1, frag_b_f16(Kb + (size_t)kk * D_, D_, 32, lane), s0);
    s1 = wmma16(aq0, frag_b_f16(Kb + (size_t)(kk + 16) * D_, D_, 0, lane), s1);
    s1 = wmma16(aq1, frag_b_f16(Kb + (size_t)(kk + 16) * D_, D_, 32, lane), s1);

    // online softmax: row m = j + 8*hl lives in register j across a 16-lane half
    float p0[8], p1[8];
    #pragma unroll
    for (int j = 0; j < 8; ++j) {
      float a = s0[j] * SCALE_, b = s1[j] * SCALE_;
      float mx = fmaxf(a, b);
      #pragma unroll
      for (int off = 8; off; off >>= 1) mx = fmaxf(mx, __shfl_xor(mx, off, 16));
      float mnew = fmaxf(mrun[j], mx);
      float corr = __expf(mrun[j] - mnew);
      mrun[j] = mnew;
      p0[j] = __expf(a - mnew);
      p1[j] = __expf(b - mnew);
      float rs = p0[j] + p1[j];
      #pragma unroll
      for (int off = 8; off; off >>= 1) rs += __shfl_xor(rs, off, 16);
      lrun[j] = lrun[j] * corr + rs;
      o[0][j] *= corr; o[1][j] *= corr; o[2][j] *= corr; o[3][j] *= corr;
    }

    // C-fragment layout -> LDS (row-major 16x32, padded to 40)
    __syncthreads();
    #pragma unroll
    for (int j = 0; j < 8; ++j) {
      int m = j + (hl << 3);
      smem[wave][m][col]      = (_Float16)p0[j];
      smem[wave][m][col + 16] = (_Float16)p1[j];
    }
    __syncthreads();

    // O += P(16x32) * V(32x64); V^T rows are contiguous over keys
    v16h ap = frag_a_f16(&smem[wave][0][0], 40, 0, lane);
    o[0] = wmma16(ap, frag_b_f16(Vb +  0 * T_, T_, kk, lane), o[0]);
    o[1] = wmma16(ap, frag_b_f16(Vb + 16 * T_, T_, kk, lane), o[1]);
    o[2] = wmma16(ap, frag_b_f16(Vb + 32 * T_, T_, kk, lane), o[2]);
    o[3] = wmma16(ap, frag_b_f16(Vb + 48 * T_, T_, kk, lane), o[3]);
  }

  // normalize and store O as f16 rows of [B_*T_][C_] per branch
  int b_ = bh >> 3, h = bh & 7;
  _Float16* Ob = Oh + (size_t)br * BRSEG +
                 ((size_t)b_ * T_ + (size_t)qt * 16) * C_ + h * D_;
  #pragma unroll
  for (int j = 0; j < 8; ++j) {
    float inv = 1.0f / lrun[j];
    int m = j + (hl << 3);
    _Float16* row = Ob + (size_t)m * C_;
    row[ 0 + col] = (_Float16)(o[0][j] * inv);
    row[16 + col] = (_Float16)(o[1][j] * inv);
    row[32 + col] = (_Float16)(o[2][j] * inv);
    row[48 + col] = (_Float16)(o[3][j] * inv);
  }
}

// ---------------------------------------------------------------------------
// Kernel 3: output projection + bias; one wave per 32x32 tile (2x2 blocking).
// ---------------------------------------------------------------------------
__global__ void oproj_kernel(const _Float16* __restrict__ Oh, OArgs oa,
                             float* __restrict__ out) {
  int wave = threadIdx.x >> 5, lane = threadIdx.x & 31;
  int tile = blockIdx.x * (blockDim.x >> 5) + wave;   // 0..4095
  int mt = tile >> 4, nt = tile & 15;
  int br = blockIdx.y;

  const _Float16* A0 = Oh + (size_t)br * BRSEG + (size_t)mt * 32 * C_;
  const _Float16* A1 = A0 + (size_t)16 * C_;
  const float*    W0 = oa.wo[br] + (size_t)nt * 32 * C_;
  const float*    W1 = W0 + (size_t)16 * C_;

  v8f cc[2][2] = {{{}, {}}, {{}, {}}};
  #pragma unroll 2
  for (int k0 = 0; k0 < C_; k0 += 32) {
    v16h a0 = frag_a_f16(A0, C_, k0, lane);
    v16h a1 = frag_a_f16(A1, C_, k0, lane);
    v16h b0 = frag_b_f32(W0, C_, k0, lane);
    v16h b1 = frag_b_f32(W1, C_, k0, lane);
    cc[0][0] = wmma16(a0, b0, cc[0][0]);
    cc[0][1] = wmma16(a0, b1, cc[0][1]);
    cc[1][0] = wmma16(a1, b0, cc[1][0]);
    cc[1][1] = wmma16(a1, b1, cc[1][1]);
  }

  int hl = lane >> 4;
  #pragma unroll
  for (int ni = 0; ni < 2; ++ni) {
    int n = nt * 32 + ni * 16 + (lane & 15);
    float bias = oa.bo[br][n];
    #pragma unroll
    for (int mi = 0; mi < 2; ++mi) {
      #pragma unroll
      for (int j = 0; j < 8; ++j) {
        int m = mt * 32 + mi * 16 + j + (hl << 3);
        out[(size_t)m * (3 * C_) + (size_t)br * C_ + n] = cc[mi][ni][j] + bias;
      }
    }
  }
}

// ---------------------------------------------------------------------------
extern "C" void kernel_launch(void* const* d_in, const int* in_sizes, int n_in,
                              void* d_out, int out_size, void* d_ws, size_t ws_size,
                              hipStream_t stream) {
  (void)in_sizes; (void)n_in; (void)out_size; (void)ws_size;

  const float* x = (const float*)d_in[0];
  _Float16* ws = (_Float16*)d_ws;
  const size_t SEG = (size_t)3 * BRSEG;     // halves per tensor (all 3 branches)
  _Float16* Qh = ws;
  _Float16* Kh = ws + SEG;
  _Float16* Vt = ws + 2 * SEG;
  _Float16* Oh = ws + 3 * SEG;              // total ws use: 4*SEG*2 = 96 MiB

  // input order: x, then per-branch {wq, wk, wv, wo, bo}
  WArgs wa;
  OArgs oa;
  for (int br = 0; br < 3; ++br) {
    wa.w[br * 3 + 0] = (const float*)d_in[1 + br * 5 + 0];
    wa.w[br * 3 + 1] = (const float*)d_in[1 + br * 5 + 1];
    wa.w[br * 3 + 2] = (const float*)d_in[1 + br * 5 + 2];
    oa.wo[br]        = (const float*)d_in[1 + br * 5 + 3];
    oa.bo[br]        = (const float*)d_in[1 + br * 5 + 4];
  }

  // 4096 wave-tiles (32x32) per (br,sel); 8 waves/block
  qkv_kernel<<<dim3(512, 9), 256, 0, stream>>>(x, wa, Qh, Kh, Vt);
  // 64 q-tiles x 64 (b,h) x 3 branches; 4 waves/block
  attn_kernel<<<dim3(16, 64, 3), 128, 0, stream>>>(Qh, Kh, Vt, Oh);
  // 4096 wave-tiles (32x32) per branch; 8 waves/block
  oproj_kernel<<<dim3(512, 3), 256, 0, stream>>>(Oh, oa, (float*)d_out);
}
